// SpatialCrossAttention_26585847562563
// MI455X (gfx1250) — matrix operand order repaired and checked by reading, not compiled, not run on hardware
//
#include <hip/hip_runtime.h>
#include <math.h>

// ---------------------------------------------------------------------------
// Problem constants (compile-time, from the reference)
// ---------------------------------------------------------------------------
#define BATCH   8
#define C_DIM   256
#define OC3     768          // 3*C
#define N_HEADS 4
#define HD      64           // head dim
#define N_POS   2304         // 48*48
#define NT      36           // N_POS / 64
#define NT2     18           // N_POS / 128

typedef __attribute__((ext_vector_type(16))) __bf16 v16bf;
typedef __attribute__((ext_vector_type(8)))  __bf16 v8bf;
typedef __attribute__((ext_vector_type(4)))  __bf16 v4bf;
typedef __attribute__((ext_vector_type(8)))  float  v8f;
typedef __attribute__((ext_vector_type(4)))  float  v4f;

// fp32 -> bf16: backend lowers to native v_cvt_pk_bf16_f32 (RNE)
__device__ __forceinline__ __bf16 f2bf(float f) {
    return static_cast<__bf16>(f);
}

__device__ __forceinline__ v8f wmma_bf16(v16bf a, v16bf b, v8f c) {
    return __builtin_amdgcn_wmma_f32_16x16x32_bf16(
        /*neg_a=*/false, a, /*neg_b=*/false, b,
        /*c_mod=*/(short)0, c, /*reuse_a=*/false, /*reuse_b=*/false);
}

// A fragment (16x32 MxK, row-major [m][k] in LDS, ld must be a multiple of 8):
// lane L (m=L%16, half=L/16) holds K = {8*half..+7} then {16+8*half..+7}
// -> two contiguous 16-byte LDS loads (ds_load_b128)
__device__ __forceinline__ v16bf load_a_frag_v(const __bf16* A, int ld, int lane) {
    int m = lane & 15, hl = lane >> 4;
    v8bf lo = *(const v8bf*)(A + m * ld + 8 * hl);
    v8bf hi = *(const v8bf*)(A + m * ld + 16 + 8 * hl);
    return __builtin_shufflevector(lo, hi, 0, 1, 2, 3, 4, 5, 6, 7,
                                   8, 9, 10, 11, 12, 13, 14, 15);
}

// B fragment (32x16 KxN) built from a TRANSPOSED tile T[n][k] (ld mult of 8):
// lane L (n=L%16, kh=L/16) holds K = {8*kh..+7} then {16+8*kh..+7}  -> two ds_load_b128
__device__ __forceinline__ v16bf load_b_fragT_v(const __bf16* T, int ld, int lane) {
    int n = lane & 15, kh = lane >> 4;
    v8bf lo = *(const v8bf*)(T + n * ld + 8 * kh);
    v8bf hi = *(const v8bf*)(T + n * ld + 16 + 8 * kh);
    return __builtin_shufflevector(lo, hi, 0, 1, 2, 3, 4, 5, 6, 7,
                                   8, 9, 10, 11, 12, 13, 14, 15);
}

// CDNA5 async global->LDS copy (ASYNCcnt-tracked), 16B per lane
__device__ __forceinline__ void async_copy_b128(unsigned lds_off, unsigned long long gaddr) {
    asm volatile("global_load_async_to_lds_b128 %0, %1, off"
                 :: "v"(lds_off), "v"(gaddr) : "memory");
}
// ASYNCcnt is per-wave and in-order: waiting <=2 guarantees everything except
// the 2 most recently issued copies (the next tile) has landed in LDS.
__device__ __forceinline__ void wait_async_le2() {
    asm volatile("s_wait_asynccnt 0x2" ::: "memory");
}
__device__ __forceinline__ void wait_async0() {
    asm volatile("s_wait_asynccnt 0x0" ::: "memory");
}

// ---------------------------------------------------------------------------
// Kernel 1: GroupNorm (1 group) statistics per batch -> mean[b], rstd[b]
// ---------------------------------------------------------------------------
__global__ __launch_bounds__(256) void gn_stats_kernel(const float* __restrict__ x,
                                                       float* __restrict__ stats) {
    const int b = blockIdx.x;
    const int tid = threadIdx.x;
    const float* xb = x + (size_t)b * C_DIM * N_POS;
    float s = 0.f, s2 = 0.f;
    for (int i = tid; i < (C_DIM * N_POS) / 4; i += 256) {
        v4f v = *(const v4f*)(xb + 4 * i);
#pragma unroll
        for (int e = 0; e < 4; ++e) { s += v[e]; s2 += v[e] * v[e]; }
    }
    __shared__ float rs[256], rs2[256];
    rs[tid] = s; rs2[tid] = s2;
    __syncthreads();
    for (int off = 128; off > 0; off >>= 1) {
        if (tid < off) { rs[tid] += rs[tid + off]; rs2[tid] += rs2[tid + off]; }
        __syncthreads();
    }
    if (tid == 0) {
        const float inv = 1.0f / (float)(C_DIM * N_POS);
        float mean = rs[0] * inv;
        float var  = rs2[0] * inv - mean * mean;
        stats[b]         = mean;
        stats[BATCH + b] = rsqrtf(var + 1e-5f);
    }
}

// ---------------------------------------------------------------------------
// Kernel 2: bias[b][o] = sum_c w_qkv[o][c] * (gn_b[c] - mean*rstd*gn_w[c])
// ---------------------------------------------------------------------------
__global__ __launch_bounds__(256) void qkv_bias_kernel(const float* __restrict__ w_qkv,
                                                       const float* __restrict__ gn_w,
                                                       const float* __restrict__ gn_b,
                                                       const float* __restrict__ stats,
                                                       float* __restrict__ bias) {
    const int b = blockIdx.y;
    const int o = blockIdx.x * 256 + threadIdx.x;   // 0..767
    const float mean = stats[b], rstd = stats[BATCH + b];
    float acc = 0.f;
    for (int c = 0; c < C_DIM; c += 4) {
        v4f w = *(const v4f*)&w_qkv[(size_t)o * C_DIM + c];
#pragma unroll
        for (int e = 0; e < 4; ++e)
            acc += w[e] * (gn_b[c + e] - mean * rstd * gn_w[c + e]);
    }
    bias[b * OC3 + o] = acc;
}

// ---------------------------------------------------------------------------
// Kernel 3: QKV gemm (GN folded into A).  bf16 WMMA, fp32 accum.
//           q scaled by HEAD_DIM^-0.5; q,k,v stored bf16 [B,H,N,64].
// ---------------------------------------------------------------------------
__global__ __launch_bounds__(256) void qkv_gemm_kernel(
    const float* __restrict__ x, const float* __restrict__ w_qkv,
    const float* __restrict__ gn_w, const float* __restrict__ stats,
    const float* __restrict__ bias,
    __bf16* __restrict__ qbuf, __bf16* __restrict__ kbuf, __bf16* __restrict__ vbuf) {
    const int b = blockIdx.z;
    const int otile = blockIdx.y * 64;
    const int ntile = blockIdx.x * 64;
    const int tid = threadIdx.x;
    const int lane = tid & 31;
    const int wave = tid >> 5;       // 8 waves: 4 (o) x 2 (n)
    const int wm = wave & 3;
    const int wn = wave >> 2;

    __shared__ __align__(16) __bf16 As[64][40];    // [o][c], GN-scaled weights
    __shared__ __align__(16) __bf16 Bst[64][40];   // TRANSPOSED x tile: [n][c]

    const float rstd = stats[BATCH + b];

    v8f acc0 = {}, acc1 = {};
    for (int kk = 0; kk < C_DIM; kk += 32) {
        __syncthreads();
        // A tile: 64x32, vectorized fp32 loads, packed 4xbf16 stores
        for (int i = tid; i < 64 * 8; i += 256) {
            int mo = i >> 3, c4 = (i & 7) * 4;
            v4f w = *(const v4f*)&w_qkv[(size_t)(otile + mo) * C_DIM + kk + c4];
            v4bf p;
#pragma unroll
            for (int e = 0; e < 4; ++e)
                p[e] = f2bf(w[e] * (gn_w[kk + c4 + e] * rstd));
            *(v4bf*)&As[mo][c4] = p;
        }
        // B tile: 32x64 of x, stored transposed [n][k]
        for (int i = tid; i < 32 * 16; i += 256) {
            int k = i >> 4, n4 = (i & 15) * 4;
            v4f v = *(const v4f*)&x[((size_t)b * C_DIM + kk + k) * N_POS + ntile + n4];
#pragma unroll
            for (int e = 0; e < 4; ++e) Bst[n4 + e][k] = f2bf(v[e]);
        }
        __syncthreads();
        v16bf a  = load_a_frag_v(&As[wm * 16][0], 40, lane);
        v16bf b0 = load_b_fragT_v(&Bst[wn * 32][0], 40, lane);
        v16bf b1 = load_b_fragT_v(&Bst[wn * 32 + 16][0], 40, lane);
        acc0 = wmma_bf16(a, b0, acc0);
        acc1 = wmma_bf16(a, b1, acc1);
    }

    // epilogue: 8 consecutive o per lane -> 8 consecutive d -> one b128 store
    const int hl = lane >> 4;
    const int n0 = ntile + wn * 32 + (lane & 15);
    const int obase = otile + wm * 16 + 8 * hl;      // multiple of 8
    const int qi = obase >> 8;                       // 0=q 1=k 2=v (constant over r)
    const int rem = obase & 255;
    const int h = rem >> 6, d0 = rem & 63;
    const float qscale = (qi == 0) ? 0.125f : 1.0f;
    __bf16* dst = (qi == 0) ? qbuf : (qi == 1 ? kbuf : vbuf);
    const size_t bh = (size_t)(b * N_HEADS + h);
    v8bf out0, out1;
#pragma unroll
    for (int r = 0; r < 8; ++r) {
        float bi = bias[b * OC3 + obase + r];
        out0[r] = f2bf((acc0[r] + bi) * qscale);
        out1[r] = f2bf((acc1[r] + bi) * qscale);
    }
    *(v8bf*)&dst[(bh * N_POS + n0) * HD + d0]        = out0;
    *(v8bf*)&dst[(bh * N_POS + n0 + 16) * HD + d0]   = out1;
}

// ---------------------------------------------------------------------------
// Kernel 4: flash attention.  One block = 128 queries of one (b,h);
//           8 waves, each owns 16 query rows over the full key stream.
//           K tiles stream through a double-buffered LDS slab filled by
//           GLOBAL_LOAD_ASYNC_TO_LDS_B128; the DMA of tile t+1 overlaps the
//           whole compute of tile t (s_wait_asynccnt 2 pipeline).
// ---------------------------------------------------------------------------
__global__ __launch_bounds__(256) void attn_kernel(
    const __bf16* __restrict__ qbuf, const __bf16* __restrict__ kbuf,
    const __bf16* __restrict__ vbuf, const int* __restrict__ mask,
    __bf16* __restrict__ attnout) {
    const int bh = blockIdx.y;
    const int b = bh >> 2;
    const int h = bh & 3;
    const int qtile = blockIdx.x * 128;
    const int tid = threadIdx.x;
    const int lane = tid & 31;
    const int wave = tid >> 5;       // 0..7 : 16 query rows each
    const int nn = lane & 15;
    const int hl = lane >> 4;

    __shared__ __align__(16) __bf16 Qs[128][72];      // [q][d]
    __shared__ __align__(16) __bf16 Ks[2][64][72];    // ping-pong [key][d]
    __shared__ __align__(16) __bf16 VsT[64][72];      // [d][key] (= B^T for P*V)
    __shared__ __align__(16) __bf16 Ps[8][16][72];    // per-wave P restage
    __shared__ int ms[64];

    const size_t bhbase = (size_t)bh * N_POS;

    // issue async DMA for K tile 0 into buffer 0 (overlaps Q staging)
    const int row0 = tid >> 3, seg0 = tid & 7;
#pragma unroll
    for (int it = 0; it < 2; ++it) {
        int row = row0 + it * 32;
        async_copy_b128(
            (unsigned)(unsigned long long)(const void*)&Ks[0][row][seg0 * 8],
            (unsigned long long)&kbuf[(bhbase + row) * HD + seg0 * 8]);
    }

    // Q tile: vector copy global->LDS (128 rows x 8 granules)
#pragma unroll
    for (int it = 0; it < 4; ++it) {
        int i = it * 256 + tid;
        int q = i >> 3, seg = i & 7;
        *(v8bf*)&Qs[q][seg * 8] =
            *(const v8bf*)&qbuf[(bhbase + qtile + q) * HD + seg * 8];
    }
    __syncthreads();

    // query fragments are loop-invariant
    v16bf qa0 = load_a_frag_v(&Qs[wave * 16][0], 72, lane);
    v16bf qa1 = load_a_frag_v(&Qs[wave * 16][32], 72, lane);

    float m_r[8], l_r[8];
    v8f O[4];
#pragma unroll
    for (int r = 0; r < 8; ++r) { m_r[r] = -3.0e38f; l_r[r] = 0.f; }
#pragma unroll
    for (int j = 0; j < 4; ++j) O[j] = (v8f){};

    int p = 0;
    for (int kt = 0; kt < N_POS; kt += 64) {
        __syncthreads();   // all waves finished reading Ks[p^1] and VsT
        const bool have_next = (kt + 64 < N_POS);
        if (have_next) {   // issue next K tile into the other buffer
#pragma unroll
            for (int it = 0; it < 2; ++it) {
                int row = row0 + it * 32;
                async_copy_b128(
                    (unsigned)(unsigned long long)(const void*)&Ks[p ^ 1][row][seg0 * 8],
                    (unsigned long long)&kbuf[(bhbase + kt + 64 + row) * HD + seg0 * 8]);
            }
        }
        // V tile: vector global load, transposed scatter into VsT
#pragma unroll
        for (int it = 0; it < 2; ++it) {
            int row = row0 + it * 32;
            v8bf vv = *(const v8bf*)&vbuf[(bhbase + kt + row) * HD + seg0 * 8];
#pragma unroll
            for (int e = 0; e < 8; ++e) VsT[seg0 * 8 + e][row] = vv[e];
        }
        if (tid < 64) {
            ms[tid] = mask[b * N_POS + kt + tid];
            if (have_next)   // prefetch next V tile (global_prefetch_b8)
                __builtin_prefetch(&vbuf[(bhbase + kt + 64 + tid) * HD], 0, 0);
        }
        // tile kt's copies (issued one iteration ago) must be in LDS;
        // the 2 copies just issued for kt+64 may remain in flight
        if (have_next) wait_async_le2(); else wait_async0();
        __syncthreads();

        // S = Q * K^T  (16 rows x 64 keys per wave), fp32 accum
        v8f S[4];
#pragma unroll
        for (int j = 0; j < 4; ++j) {
            v16bf b0 = load_b_fragT_v(&Ks[p][j * 16][0], 72, lane);
            v16bf b1 = load_b_fragT_v(&Ks[p][j * 16][32], 72, lane);
            v8f s = {};
            s = wmma_bf16(qa0, b0, s);
            s = wmma_bf16(qa1, b1, s);
            if (ms[j * 16 + nn] == 0) {      // key mask
#pragma unroll
                for (int r = 0; r < 8; ++r) s[r] = -3.0e38f;
            }
            S[j] = s;
        }

        // online softmax; rows live in lane halves -> 16-lane xor reductions
#pragma unroll
        for (int r = 0; r < 8; ++r) {
            float mx = fmaxf(fmaxf(S[0][r], S[1][r]), fmaxf(S[2][r], S[3][r]));
#pragma unroll
            for (int off = 1; off < 16; off <<= 1)
                mx = fmaxf(mx, __shfl_xor(mx, off, 32));
            float mnew = fmaxf(m_r[r], mx);
            float alpha = __expf(m_r[r] - mnew);
            m_r[r] = mnew;
            float rs = 0.f;
#pragma unroll
            for (int j = 0; j < 4; ++j) {
                float ppp = __expf(S[j][r] - mnew);
                S[j][r] = ppp;
                rs += ppp;
            }
#pragma unroll
            for (int off = 1; off < 16; off <<= 1)
                rs += __shfl_xor(rs, off, 32);
            l_r[r] = l_r[r] * alpha + rs;
#pragma unroll
            for (int j = 0; j < 4; ++j) O[j][r] *= alpha;
        }

        // restage P (C-layout -> A-layout) through per-wave LDS slab.
        // LDS ops from one wave are in-order (ISA 7.3), and the slab is
        // wave-private, so no block barrier is required before re-reading.
#pragma unroll
        for (int j = 0; j < 4; ++j)
#pragma unroll
            for (int r = 0; r < 8; ++r)
                Ps[wave][r + 8 * hl][j * 16 + nn] = f2bf(S[j][r]);

        // O += P * V
        v16bf pa0 = load_a_frag_v(&Ps[wave][0][0], 72, lane);
        v16bf pa1 = load_a_frag_v(&Ps[wave][0][32], 72, lane);
#pragma unroll
        for (int j = 0; j < 4; ++j) {
            v16bf b0 = load_b_fragT_v(&VsT[j * 16][0], 72, lane);
            v16bf b1 = load_b_fragT_v(&VsT[j * 16][32], 72, lane);
            O[j] = wmma_bf16(pa0, b0, O[j]);
            O[j] = wmma_bf16(pa1, b1, O[j]);
        }
        p ^= 1;
    }

    // normalize; 8 consecutive q per lane are contiguous along N -> b128 stores
    float inv_r[8];
#pragma unroll
    for (int r = 0; r < 8; ++r) inv_r[r] = 1.0f / fmaxf(l_r[r], 1e-30f);
    const int qbase = qtile + wave * 16 + 8 * hl;    // multiple of 8
#pragma unroll
    for (int j = 0; j < 4; ++j) {
        int d = j * 16 + nn;
        v8bf ov;
#pragma unroll
        for (int r = 0; r < 8; ++r)
            ov[r] = f2bf(O[j][r] * inv_r[r]);
        *(v8bf*)&attnout[((size_t)b * C_DIM + h * HD + d) * N_POS + qbase] = ov;
    }
}

// ---------------------------------------------------------------------------
// Kernel 5: output projection + bias + residual, fp32 out
// ---------------------------------------------------------------------------
__global__ __launch_bounds__(256) void proj_kernel(
    const float* __restrict__ x, const float* __restrict__ w_proj,
    const float* __restrict__ b_proj, const __bf16* __restrict__ attnout,
    float* __restrict__ out) {
    const int b = blockIdx.z;
    const int otile = blockIdx.y * 64;
    const int ntile = blockIdx.x * 64;
    const int tid = threadIdx.x;
    const int lane = tid & 31;
    const int wave = tid >> 5;
    const int wm = wave & 3;
    const int wn = wave >> 2;

    __shared__ __align__(16) __bf16 As[64][40];    // [o][c]
    __shared__ __align__(16) __bf16 Bst[64][40];   // TRANSPOSED attnout tile [n][c]

    v8f acc0 = {}, acc1 = {};
    for (int kk = 0; kk < C_DIM; kk += 32) {
        __syncthreads();
        for (int i = tid; i < 64 * 8; i += 256) {
            int mo = i >> 3, c4 = (i & 7) * 4;
            v4f w = *(const v4f*)&w_proj[(size_t)(otile + mo) * C_DIM + kk + c4];
            v4bf p;
#pragma unroll
            for (int e = 0; e < 4; ++e) p[e] = f2bf(w[e]);
            *(v4bf*)&As[mo][c4] = p;
        }
        for (int i = tid; i < 32 * 8; i += 256) {
            int k = i >> 3, n8 = (i & 7) * 8;
            v8bf v = *(const v8bf*)&attnout[((size_t)b * C_DIM + kk + k) * N_POS + ntile + n8];
#pragma unroll
            for (int e = 0; e < 8; ++e) Bst[n8 + e][k] = v[e];
        }
        __syncthreads();
        v16bf a  = load_a_frag_v(&As[wm * 16][0], 40, lane);
        v16bf b0 = load_b_fragT_v(&Bst[wn * 32][0], 40, lane);
        v16bf b1 = load_b_fragT_v(&Bst[wn * 32 + 16][0], 40, lane);
        acc0 = wmma_bf16(a, b0, acc0);
        acc1 = wmma_bf16(a, b1, acc1);
    }

    const int hl = lane >> 4;
    const int n0 = ntile + wn * 32 + (lane & 15);
#pragma unroll
    for (int r = 0; r < 8; ++r) {
        int o = otile + wm * 16 + r + 8 * hl;
        size_t base = ((size_t)b * C_DIM + o) * N_POS;
        float bp = b_proj[o];
        out[base + n0]      = x[base + n0]      + bp + acc0[r];
        out[base + n0 + 16] = x[base + n0 + 16] + bp + acc1[r];
    }
}

// ---------------------------------------------------------------------------
// Launch
// ---------------------------------------------------------------------------
extern "C" void kernel_launch(void* const* d_in, const int* in_sizes, int n_in,
                              void* d_out, int out_size, void* d_ws, size_t ws_size,
                              hipStream_t stream) {
    (void)in_sizes; (void)n_in; (void)out_size; (void)ws_size;
    const float* x      = (const float*)d_in[0];
    const int*   mask   = (const int*)d_in[1];
    const float* gn_w   = (const float*)d_in[2];
    const float* gn_b   = (const float*)d_in[3];
    const float* w_qkv  = (const float*)d_in[4];
    const float* w_proj = (const float*)d_in[5];
    const float* b_proj = (const float*)d_in[6];
    float* out = (float*)d_out;

    // workspace layout (~37.8 MB): stats | bias | q | k | v | attnout
    char* ws = (char*)d_ws;
    float* stats = (float*)ws;                           // 16 floats (pad to 256 B)
    float* bias  = (float*)(ws + 256);                   // 8*768 floats
    const size_t qkv_bytes = (size_t)BATCH * N_HEADS * N_POS * HD * sizeof(__bf16);
    __bf16* qbuf    = (__bf16*)(ws + 256 + (size_t)BATCH * OC3 * sizeof(float));
    __bf16* kbuf    = (__bf16*)((char*)qbuf + qkv_bytes);
    __bf16* vbuf    = (__bf16*)((char*)kbuf + qkv_bytes);
    __bf16* attnout = (__bf16*)((char*)vbuf + qkv_bytes);

    gn_stats_kernel<<<BATCH, 256, 0, stream>>>(x, stats);
    qkv_bias_kernel<<<dim3(3, BATCH), 256, 0, stream>>>(w_qkv, gn_w, gn_b, stats, bias);
    qkv_gemm_kernel<<<dim3(NT, OC3 / 64, BATCH), 256, 0, stream>>>(
        x, w_qkv, gn_w, stats, bias, qbuf, kbuf, vbuf);
    attn_kernel<<<dim3(NT2, BATCH * N_HEADS), 256, 0, stream>>>(
        qbuf, kbuf, vbuf, mask, attnout);
    proj_kernel<<<dim3(NT, C_DIM / 64, BATCH), 256, 0, stream>>>(
        x, w_proj, b_proj, attnout, out);
}
